// DGCNN_70643622084664
// MI455X (gfx1250) — compile-verified, matrix operand order checked by reading, not run on hardware
//
#include <hip/hip_runtime.h>
#include <cstdint>

#define EPS_   1e-5f
#define SLOPE_ 0.2f

typedef __attribute__((ext_vector_type(16))) __bf16 v16bf;
typedef __attribute__((ext_vector_type(8)))  float  v8f;

union FragU { v16bf v; unsigned short u[16]; };

#define V8F_ZERO {0.f,0.f,0.f,0.f,0.f,0.f,0.f,0.f}

__device__ __forceinline__ unsigned short f2bf(float f) {
  unsigned int u = __float_as_uint(f);
  u += 0x7FFFu + ((u >> 16) & 1u);          // round-to-nearest-even to bf16
  return (unsigned short)(u >> 16);
}
__device__ __forceinline__ float leaky(float v) { return v >= 0.f ? v : SLOPE_ * v; }

// monotone float -> uint key for atomic max pooling
__device__ __forceinline__ unsigned int fkey(float f) {
  unsigned int u = __float_as_uint(f);
  return (u & 0x80000000u) ? ~u : (u | 0x80000000u);
}
__device__ __forceinline__ float funkey(unsigned int k) {
  unsigned int u = (k & 0x80000000u) ? (k & 0x7FFFFFFFu) : ~k;
  return __uint_as_float(u);
}

// Load one 16x32 bf16 operand fragment (A layout per CDNA5 ISA 7.12.2):
// lane l: row = l&15, half = l>>4; elems 0..7 -> K = half*8+i, elems 8..15 -> K = 16+half*8+i.
// Works for both A tiles and B tiles stored as W row-major (co, ci) == B^T.
__device__ __forceinline__ v16bf frag_load(const unsigned short* tile, int stride) {
  int l = threadIdx.x & 31;
  int r = l & 15, hh = (l >> 4) & 1;
  const unsigned short* p = tile + r * stride + hh * 8;
  FragU f;
#pragma unroll
  for (int i = 0; i < 8; ++i) { f.u[i] = p[i]; f.u[8 + i] = p[16 + i]; }
  return f.v;
}

__device__ __forceinline__ v8f wmma_bf16(v16bf a, v16bf b, v8f c) {
  return __builtin_amdgcn_wmma_f32_16x16x32_bf16(false, a, false, b, (short)0, c, false, false);
}

// ---------------------------------------------------------------------------
// K0: fold scale into weights, convert to bf16 with K/N padding, zero pool keys
// ---------------------------------------------------------------------------
struct PrepArgs {
  const float *W2a,*b2a,*g2a,*be2a;   // conv2 L1: (104,44)
  const float *W2b,*b2b,*g2b,*be2b;   // conv2 L2: (256,104)
  const float *W3a,*b3a,*g3a,*be3a;   // conv3 L1: (266,278)
  const float *W3b,*b3b,*g3b,*be3b;   // conv3 L2: (256,266)
  unsigned short *oW2a,*oW2b,*oW3a,*oW3b;   // (112,64) (256,128) (272,288) (256,288)
  float *ot2a,*ot2b,*ot3a,*ot3b;
  unsigned int* keys;                 // 16*256 pooled-max keys
};

__global__ void __launch_bounds__(256) prep_kernel(PrepArgs a) {
  int t = blockIdx.x * blockDim.x + threadIdx.x;
  int stride = gridDim.x * blockDim.x;
  const float inv = rsqrtf(1.f + EPS_);
  for (int i = t; i < 112 * 64; i += stride) {
    int o = i >> 6, c = i & 63;
    float v = (o < 104 && c < 44) ? a.g2a[o] * inv * a.W2a[o * 44 + c] : 0.f;
    a.oW2a[i] = f2bf(v);
  }
  for (int i = t; i < 112; i += stride)
    a.ot2a[i] = (i < 104) ? a.g2a[i] * inv * a.b2a[i] + a.be2a[i] : 0.f;
  for (int i = t; i < 256 * 128; i += stride) {
    int o = i >> 7, c = i & 127;
    float v = (c < 104) ? a.g2b[o] * inv * a.W2b[o * 104 + c] : 0.f;
    a.oW2b[i] = f2bf(v);
  }
  for (int i = t; i < 256; i += stride)
    a.ot2b[i] = a.g2b[i] * inv * a.b2b[i] + a.be2b[i];
  for (int i = t; i < 272 * 288; i += stride) {
    int o = i / 288, c = i % 288;
    float v = (o < 266 && c < 278) ? a.g3a[o] * inv * a.W3a[o * 278 + c] : 0.f;
    a.oW3a[i] = f2bf(v);
  }
  for (int i = t; i < 272; i += stride)
    a.ot3a[i] = (i < 266) ? a.g3a[i] * inv * a.b3a[i] + a.be3a[i] : 0.f;
  for (int i = t; i < 256 * 288; i += stride) {
    int o = i / 288, c = i % 288;
    float v = (c < 266) ? a.g3b[o] * inv * a.W3b[o * 266 + c] : 0.f;
    a.oW3b[i] = f2bf(v);
  }
  for (int i = t; i < 256; i += stride)
    a.ot3b[i] = a.g3b[i] * inv * a.b3b[i] + a.be3b[i];
  for (int i = t; i < 16 * 256; i += stride) a.keys[i] = 0u;
}

// ---------------------------------------------------------------------------
// K1/K3: pairwise pd = 2<xi,xj> - |xi|^2 - |xj|^2, top-10 (largest) per row.
// One wave32 per row; distance row cached in LDS; 10 argmax sweeps + shuffles.
// ---------------------------------------------------------------------------
template <int C>
__global__ void __launch_bounds__(256) topk_kernel(const float* __restrict__ xp,
                                                   int sb, int sc, int sn,
                                                   int* __restrict__ idx_out) {
  __shared__ float sd[8][2048];   // 64 KB: one 2048-row per wave
  int w = threadIdx.x >> 5, l = threadIdx.x & 31;
  int row = blockIdx.x * 8 + w;
  int b = row >> 11, n = row & 2047;
  const float* xb = xp + (size_t)b * sb;
  float xc[C];
  float sqi = 0.f;
#pragma unroll
  for (int c = 0; c < C; ++c) { xc[c] = xb[c * sc + n * sn]; sqi += xc[c] * xc[c]; }
  for (int j = l; j < 2048; j += 32) {
    float dot = 0.f, sqj = 0.f;
#pragma unroll
    for (int c = 0; c < C; ++c) { float v = xb[c * sc + j * sn]; dot += xc[c] * v; sqj += v * v; }
    sd[w][j] = 2.f * dot - sqi - sqj;
  }
  __syncthreads();
  for (int it = 0; it < 10; ++it) {
    float m = -3.0e38f; int mi = 0;
    for (int j = l; j < 2048; j += 32) {
      float v = sd[w][j];
      if (v > m) { m = v; mi = j; }
    }
#pragma unroll
    for (int off = 16; off > 0; off >>= 1) {
      float om = __shfl_xor(m, off, 32);
      int   oi = __shfl_xor(mi, off, 32);
      if (om > m || (om == m && oi < mi)) { m = om; mi = oi; }
    }
    if (l == 0) { idx_out[row * 10 + it] = mi; sd[w][mi] = -3.0e38f; }
    __syncthreads();
  }
}

// ---------------------------------------------------------------------------
// K2: conv1 stack (4->9->22) fused with edge-feature build and max over k.
// Tiny channel counts -> plain VALU FMA, one thread per point.
// ---------------------------------------------------------------------------
__global__ void __launch_bounds__(256) conv1_fused(
    const float* __restrict__ x, const int* __restrict__ idx,
    const float* __restrict__ W1, const float* __restrict__ b1,
    const float* __restrict__ g1, const float* __restrict__ be1,
    const float* __restrict__ W2, const float* __restrict__ b2,
    const float* __restrict__ g2, const float* __restrict__ be2,
    float* __restrict__ x1) {
  int p = blockIdx.x * blockDim.x + threadIdx.x;   // 32768 points
  int b = p >> 11, n = p & 2047;
  const float inv = rsqrtf(1.f + EPS_);
  const float* xb = x + (size_t)b * 4096;
  float c0 = xb[n], c1 = xb[2048 + n];
  float best[22];
#pragma unroll
  for (int o = 0; o < 22; ++o) best[o] = -3.0e38f;
  for (int kk = 0; kk < 10; ++kk) {
    int j = idx[p * 10 + kk];
    float e0 = xb[j] - c0, e1 = xb[2048 + j] - c1, e2 = c0, e3 = c1;
    float h[9];
#pragma unroll
    for (int o = 0; o < 9; ++o) {
      float a = W1[o*4]*e0 + W1[o*4+1]*e1 + W1[o*4+2]*e2 + W1[o*4+3]*e3 + b1[o];
      h[o] = leaky(g1[o] * inv * a + be1[o]);
    }
#pragma unroll
    for (int o = 0; o < 22; ++o) {
      float a = b2[o];
#pragma unroll
      for (int i = 0; i < 9; ++i) a += W2[o*9+i] * h[i];
      best[o] = fmaxf(best[o], leaky(g2[o] * inv * a + be2[o]));
    }
  }
  float* op = x1 + (size_t)p * 22;   // point-major (b,n,c)
#pragma unroll
  for (int o = 0; o < 22; ++o) op[o] = best[o];
}

// ---------------------------------------------------------------------------
// K4: conv2 stack (44->104->256) via bf16 WMMA. One wave per point: rows =
// 10 neighbors padded to 16, A-fragments built in registers from gathers,
// layer-1 result through LDS, max-over-k via accumulator rows + shfl_xor(16).
// ---------------------------------------------------------------------------
__global__ void __launch_bounds__(256) edgeconv2_wmma(
    const float* __restrict__ x1, const int* __restrict__ idx,
    const unsigned short* __restrict__ W1, const float* __restrict__ t1,
    const unsigned short* __restrict__ W2, const float* __restrict__ t2,
    float* __restrict__ x2) {
  __shared__ __align__(16) unsigned short sH[8][16 * 128];  // 32 KB layer-1 tiles
  int w = threadIdx.x >> 5, l = threadIdx.x & 31;
  int p = blockIdx.x * 8 + w;
  int b = p >> 11, n = p & 2047;
  int r = l & 15, hh = (l >> 4) & 1;

  // warm L2-resident bf16 weights (global_prefetch_b8)
  for (int i = threadIdx.x; i < 512; i += 256)
    __builtin_prefetch(W2 + i * 64, 0, 2);

  const float* cen = x1 + (size_t)(b * 2048 + n) * 22;
  FragU fa[2];
  if (r < 10) {
    int j = idx[p * 10 + r];
    const float* nb = x1 + (size_t)(b * 2048 + j) * 22;
#pragma unroll
    for (int kt = 0; kt < 2; ++kt)
#pragma unroll
      for (int i = 0; i < 16; ++i) {
        int c = kt * 32 + ((i < 8) ? (hh * 8 + i) : (16 + hh * 8 + (i - 8)));
        float v = (c < 22) ? (nb[c] - cen[c]) : ((c < 44) ? cen[c - 22] : 0.f);
        fa[kt].u[i] = f2bf(v);
      }
  } else {
#pragma unroll
    for (int kt = 0; kt < 2; ++kt)
#pragma unroll
      for (int i = 0; i < 16; ++i) fa[kt].u[i] = 0;
  }

  // layer 1: K 44->64, N 104->112
  for (int nt = 0; nt < 7; ++nt) {
    v8f acc = V8F_ZERO;
#pragma unroll
    for (int kt = 0; kt < 2; ++kt) {
      v16bf bmat = frag_load(W1 + nt * 16 * 64 + kt * 32, 64);
      acc = wmma_bf16(fa[kt].v, bmat, acc);
    }
    int ch = nt * 16 + r;
    float tc = t1[ch];
#pragma unroll
    for (int rr = 0; rr < 8; ++rr) {
      int m = hh * 8 + rr;
      sH[w][m * 128 + ch] = f2bf(leaky(acc[rr] + tc));
    }
  }
  for (int i = l; i < 256; i += 32)
    sH[w][(i >> 4) * 128 + 112 + (i & 15)] = 0;   // zero K-pad cols 112..127
  __syncthreads();

  // layer 2: K 104->128, N 256; rows 0..9 valid -> max over k
  float* op = x2 + (size_t)(b * 2048 + n) * 256;
  for (int nt = 0; nt < 16; ++nt) {
    v8f acc = V8F_ZERO;
#pragma unroll
    for (int kt = 0; kt < 4; ++kt) {
      v16bf amat = frag_load(&sH[w][kt * 32], 128);
      v16bf bmat = frag_load(W2 + nt * 16 * 128 + kt * 32, 128);
      acc = wmma_bf16(amat, bmat, acc);
    }
    int ch = nt * 16 + r;
    float tc = t2[ch];
    int nvalid = hh ? 2 : 8;   // half 0: rows 0..7 ; half 1: rows 8..15 (8,9 valid)
    float m = -3.0e38f;
#pragma unroll
    for (int rr = 0; rr < 8; ++rr)
      if (rr < nvalid) m = fmaxf(m, leaky(acc[rr] + tc));
    m = fmaxf(m, __shfl_xor(m, 16, 32));
    if (hh == 0) op[ch] = m;
  }
}

// ---------------------------------------------------------------------------
// K5a: conv3 layer 1 (278->288 K, 266->272 N) -> bf16 h3 in scratch.
// A fragments (9 x v16bf) resident in registers; no LDS needed.
// ---------------------------------------------------------------------------
__global__ void __launch_bounds__(256) conv3a_wmma(
    const float* __restrict__ x1, const float* __restrict__ x2,
    const unsigned short* __restrict__ W, const float* __restrict__ t,
    unsigned short* __restrict__ h3) {
  int w = threadIdx.x >> 5, l = threadIdx.x & 31;
  int tile = blockIdx.x * 8 + w;           // 2048 tiles of 16 points
  int r = l & 15, hh = (l >> 4) & 1;
  int p = tile * 16 + r;
  const float* px1 = x1 + (size_t)p * 22;
  const float* px2 = x2 + (size_t)p * 256;
  for (int i = threadIdx.x; i < 1224; i += 256)
    __builtin_prefetch(W + i * 64, 0, 2);
  FragU fa[9];
#pragma unroll
  for (int kt = 0; kt < 9; ++kt)
#pragma unroll
    for (int i = 0; i < 16; ++i) {
      int c = kt * 32 + ((i < 8) ? (hh * 8 + i) : (16 + hh * 8 + (i - 8)));
      float v = (c < 22) ? px1[c] : ((c < 278) ? px2[c - 22] : 0.f);
      fa[kt].u[i] = f2bf(v);
    }
  unsigned short* hb = h3 + (size_t)tile * (16 * 288);
  for (int nt = 0; nt < 17; ++nt) {
    v8f acc = V8F_ZERO;
#pragma unroll
    for (int kt = 0; kt < 9; ++kt) {
      v16bf bmat = frag_load(W + nt * 16 * 288 + kt * 32, 288);
      acc = wmma_bf16(fa[kt].v, bmat, acc);
    }
    int ch = nt * 16 + r;
    float tc = t[ch];
#pragma unroll
    for (int rr = 0; rr < 8; ++rr) {
      int m = hh * 8 + rr;
      hb[m * 288 + ch] = f2bf(leaky(acc[rr] + tc));
    }
  }
  for (int i = l; i < 256; i += 32)
    hb[(i >> 4) * 288 + 272 + (i & 15)] = 0;   // zero K-pad cols 272..287
}

// ---------------------------------------------------------------------------
// K5b: conv3 layer 2 (266->288 K, 256 N) + global max-pool over N via
// ordered-uint atomicMax. A fragments streamed straight from h3.
// ---------------------------------------------------------------------------
__global__ void __launch_bounds__(256) conv3b_wmma(
    const unsigned short* __restrict__ h3,
    const unsigned short* __restrict__ W, const float* __restrict__ t,
    unsigned int* __restrict__ keys) {
  int w = threadIdx.x >> 5, l = threadIdx.x & 31;
  int tile = blockIdx.x * 8 + w;
  int r = l & 15, hh = (l >> 4) & 1;
  int b = tile >> 7;   // 128 tiles per batch (2048/16)
  const unsigned short* A = h3 + (size_t)tile * (16 * 288);
  for (int i = threadIdx.x; i < 1152; i += 256)
    __builtin_prefetch(W + i * 64, 0, 2);
  FragU fa[9];
#pragma unroll
  for (int kt = 0; kt < 9; ++kt) fa[kt].v = frag_load(A + kt * 32, 288);
  for (int nt = 0; nt < 16; ++nt) {
    v8f acc = V8F_ZERO;
#pragma unroll
    for (int kt = 0; kt < 9; ++kt) {
      v16bf bmat = frag_load(W + nt * 16 * 288 + kt * 32, 288);
      acc = wmma_bf16(fa[kt].v, bmat, acc);
    }
    int ch = nt * 16 + r;
    float tc = t[ch];
    float m = -3.0e38f;
#pragma unroll
    for (int rr = 0; rr < 8; ++rr) m = fmaxf(m, leaky(acc[rr] + tc));
    m = fmaxf(m, __shfl_xor(m, 16, 32));
    if (hh == 0) atomicMax(&keys[b * 256 + ch], fkey(m));
  }
}

// ---------------------------------------------------------------------------
// K6: decode pooled keys, MLP 256->148->40 (scale folded on the fly),
// log-softmax. Tiny: one block.
// ---------------------------------------------------------------------------
__global__ void __launch_bounds__(256) mlp_head(
    const unsigned int* __restrict__ keys,
    const float* __restrict__ W1, const float* __restrict__ b1,
    const float* __restrict__ g1, const float* __restrict__ be1,
    const float* __restrict__ W2, const float* __restrict__ b2,
    float* __restrict__ out) {
  __shared__ float pooled[16 * 256];
  __shared__ float h1[16 * 148];
  __shared__ float logits[16 * 40];
  int t = threadIdx.x;
  const float inv = rsqrtf(1.f + EPS_);
  for (int i = t; i < 16 * 256; i += 256) pooled[i] = funkey(keys[i]);
  __syncthreads();
  for (int i = t; i < 16 * 148; i += 256) {
    int bi = i / 148, o = i % 148;
    float a = 0.f;
    for (int c = 0; c < 256; ++c) a += W1[o * 256 + c] * pooled[bi * 256 + c];
    a += b1[o];
    h1[i] = leaky(g1[o] * inv * a + be1[o]);
  }
  __syncthreads();
  for (int i = t; i < 16 * 40; i += 256) {
    int bi = i / 40, o = i % 40;
    float a = b2[o];
    for (int c = 0; c < 148; ++c) a += W2[o * 148 + c] * h1[bi * 148 + c];
    logits[i] = a;
  }
  __syncthreads();
  if (t < 16) {
    float mx = -3.0e38f;
    for (int o = 0; o < 40; ++o) mx = fmaxf(mx, logits[t * 40 + o]);
    float s = 0.f;
    for (int o = 0; o < 40; ++o) s += expf(logits[t * 40 + o] - mx);
    float ls = logf(s);
    for (int o = 0; o < 40; ++o) out[t * 40 + o] = logits[t * 40 + o] - mx - ls;
  }
}

// ---------------------------------------------------------------------------
extern "C" void kernel_launch(void* const* d_in, const int* in_sizes, int n_in,
                              void* d_out, int out_size, void* d_ws, size_t ws_size,
                              hipStream_t stream) {
  (void)in_sizes; (void)n_in; (void)out_size; (void)ws_size;
  const float* x = (const float*)d_in[0];
  const float *c1W1=(const float*)d_in[1],  *c1b1=(const float*)d_in[2],
              *c1g1=(const float*)d_in[3],  *c1be1=(const float*)d_in[4];
  const float *c1W2=(const float*)d_in[5],  *c1b2=(const float*)d_in[6],
              *c1g2=(const float*)d_in[7],  *c1be2=(const float*)d_in[8];
  const float *c2W1=(const float*)d_in[9],  *c2b1=(const float*)d_in[10],
              *c2g1=(const float*)d_in[11], *c2be1=(const float*)d_in[12];
  const float *c2W2=(const float*)d_in[13], *c2b2=(const float*)d_in[14],
              *c2g2=(const float*)d_in[15], *c2be2=(const float*)d_in[16];
  const float *c3W1=(const float*)d_in[17], *c3b1=(const float*)d_in[18],
              *c3g1=(const float*)d_in[19], *c3be1=(const float*)d_in[20];
  const float *c3W2=(const float*)d_in[21], *c3b2=(const float*)d_in[22],
              *c3g2=(const float*)d_in[23], *c3be2=(const float*)d_in[24];
  const float *lW1=(const float*)d_in[25],  *lb1=(const float*)d_in[26],
              *lg1=(const float*)d_in[27],  *lbe1=(const float*)d_in[28];
  const float *lW2=(const float*)d_in[29],  *lb2=(const float*)d_in[30];
  float* out = (float*)d_out;

  char* ws = (char*)d_ws;
  size_t off = 0;
  auto take = [&](size_t bytes) -> void* {
    void* pp = (void*)(ws + off);
    off = (off + bytes + 255) & ~(size_t)255;
    return pp;
  };
  int*            idx1 = (int*)take((size_t)32768 * 10 * 4);
  int*            idx2 = (int*)take((size_t)32768 * 10 * 4);
  float*          x1   = (float*)take((size_t)32768 * 22 * 4);
  float*          x2   = (float*)take((size_t)32768 * 256 * 4);
  unsigned short* h3   = (unsigned short*)take((size_t)32768 * 288 * 2);
  unsigned short* W2a  = (unsigned short*)take(112 * 64 * 2);
  float*          t2a  = (float*)take(112 * 4);
  unsigned short* W2b  = (unsigned short*)take(256 * 128 * 2);
  float*          t2b  = (float*)take(256 * 4);
  unsigned short* W3a  = (unsigned short*)take(272 * 288 * 2);
  float*          t3a  = (float*)take(272 * 4);
  unsigned short* W3b  = (unsigned short*)take(256 * 288 * 2);
  float*          t3b  = (float*)take(256 * 4);
  unsigned int*   keys = (unsigned int*)take(16 * 256 * 4);

  PrepArgs pa;
  pa.W2a = c2W1; pa.b2a = c2b1; pa.g2a = c2g1; pa.be2a = c2be1;
  pa.W2b = c2W2; pa.b2b = c2b2; pa.g2b = c2g2; pa.be2b = c2be2;
  pa.W3a = c3W1; pa.b3a = c3b1; pa.g3a = c3g1; pa.be3a = c3be1;
  pa.W3b = c3W2; pa.b3b = c3b2; pa.g3b = c3g2; pa.be3b = c3be2;
  pa.oW2a = W2a; pa.oW2b = W2b; pa.oW3a = W3a; pa.oW3b = W3b;
  pa.ot2a = t2a; pa.ot2b = t2b; pa.ot3a = t3a; pa.ot3b = t3b;
  pa.keys = keys;

  prep_kernel<<<256, 256, 0, stream>>>(pa);
  topk_kernel<2><<<4096, 256, 0, stream>>>(x, 4096, 2048, 1, idx1);
  conv1_fused<<<128, 256, 0, stream>>>(x, idx1,
      c1W1, c1b1, c1g1, c1be1, c1W2, c1b2, c1g2, c1be2, x1);
  topk_kernel<22><<<4096, 256, 0, stream>>>(x1, 2048 * 22, 1, 22, idx2);
  edgeconv2_wmma<<<4096, 256, 0, stream>>>(x1, idx2, W2a, t2a, W2b, t2b, x2);
  conv3a_wmma<<<256, 256, 0, stream>>>(x1, x2, W3a, t3a, h3);
  conv3b_wmma<<<256, 256, 0, stream>>>(h3, W3b, t3b, keys);
  mlp_head<<<1, 256, 0, stream>>>(keys, lW1, lb1, lg1, lbe1, lW2, lb2, out);
}
